// MultimodalEmotionModel_Refactored_67405216743465
// MI455X (gfx1250) — compile-verified, matrix operand order checked by reading, not run on hardware
//
#include <hip/hip_runtime.h>
#include <hip/hip_bf16.h>
#include <math.h>

// Problem constants (from reference)
#define B_    16
#define T_    256
#define H_    512
#define P_    768
#define L_    4
#define M_    3
#define DIN_  1024     // H * EXPAND
#define N_    64
#define DCONV_ 4
#define DTR_  32
#define XPW_  160      // DTR + 2*N
#define EPS_  1e-5f

typedef __attribute__((ext_vector_type(16))) __bf16 v16bf;
typedef __attribute__((ext_vector_type(8)))  float  v8f;

__device__ __forceinline__ unsigned pack_bf16(float x, float y) {
  __hip_bfloat162 h = __float22bfloat162_rn(make_float2(x, y));
  unsigned u; __builtin_memcpy(&u, &h, 4);
  return u;
}
__device__ __forceinline__ float sigmoidf_(float x){ return 1.0f/(1.0f+__expf(-x)); }
__device__ __forceinline__ float siluf_(float x){ return x*sigmoidf_(x); }
__device__ __forceinline__ float geluf_(float x){
  const float c = 0.7978845608028654f;
  return 0.5f*x*(1.0f + tanhf(c*(x + 0.044715f*x*x*x)));
}
__device__ __forceinline__ float softplusf_(float x){
  return (x > 20.0f) ? x : log1pf(__expf(x));
}

// Async global->LDS copy of 16 bytes per lane (CDNA5, tracked by ASYNCcnt).
__device__ __forceinline__ void async_b128_to_lds(unsigned lds_off, const void* g) {
  asm volatile("global_load_async_to_lds_b128 %0, %1, off"
               :: "v"(lds_off), "v"((unsigned long long)(size_t)g) : "memory");
}

// Convert a weight slice fp32 -> bf16 (vectorized 4-wide).
__global__ __launch_bounds__(256)
void wcvt_kernel(const float* __restrict__ src, unsigned short* __restrict__ dst, int n4)
{
  int i = blockIdx.x * blockDim.x + threadIdx.x;   // index of 4-element group
  if (i >= n4) return;
  float4 v = ((const float4*)src)[i];
  ((uint2*)dst)[i] = make_uint2(pack_bf16(v.x, v.y), pack_bf16(v.z, v.w));
}

// ---------------------------------------------------------------------------
// Generic GEMM: C[row,col] = epi( sum_k A[row*lda+k] * W[col*K+k] + bias[col] )
//               * (mask ? mask[row] : 1)
// A: fp32 activations (row-major, lda). Wbf: bf16 weights (Ncols x K row-major).
// Block: 128 threads = 4 waves; block tile 128(M) x 64(N); each wave 32x64
// (8 accumulators => 8 WMMAs per K-step). K tiled by 32 with DOUBLE-BUFFERED
// LDS tiles: B via global_load_async_to_lds_b128 pipelined with
// s_wait_asynccnt 0x2 (tile it+1 in flight while computing tile it);
// A via b128 loads + packed bf16 cvt + ds_store.
// EPI: 0=none, 1=+bias, 2=gelu(x+bias), 3=softplus(x+bias)
// ---------------------------------------------------------------------------
template<int EPI>
__global__ __launch_bounds__(128)
void gemm_wmma_bf16(const float* __restrict__ A, const unsigned short* __restrict__ Wbf,
                    float* __restrict__ C, int Ncols, int K, int lda, int ldc,
                    const float* __restrict__ bias, const float* __restrict__ mask)
{
  __shared__ __align__(16) unsigned short As[2][128*36]; // 128 rows x 32 k, stride 36
  __shared__ __align__(16) unsigned short Bs[2][64*40];  // 64 cols x 32 k, stride 40
  const int tid  = threadIdx.x;
  const int wid  = tid >> 5;
  const int lane = tid & 31;
  const int row0 = blockIdx.y * 128;
  const int col0 = blockIdx.x * 64;
  const int ntiles = K >> 5;

  v8f acc[8];
  #pragma unroll
  for (int i = 0; i < 8; ++i) acc[i] = (v8f){};

  // B tile: 64x32 bf16 = 4KB; 2 async b128 chunks per thread per tile.
  auto stageB = [&](int it, int buf) {
    int k0 = it << 5;
    #pragma unroll
    for (int q = 0; q < 2; ++q) {
      int idx = q*128 + tid;             // 256 chunks of 8 bf16
      int c   = idx >> 2;
      int kc  = (idx & 3) * 8;
      int col = col0 + c; if (col >= Ncols) col = Ncols - 1;  // clamp; store-guarded
      unsigned loff = (unsigned)(size_t)(const void*)&Bs[buf][c*40 + kc];
      async_b128_to_lds(loff, Wbf + (size_t)col*K + (k0 + kc));
    }
  };
  // A tile: 128x32 fp32 -> bf16; 8 float4 chunks per thread per tile.
  auto stageA = [&](int it, int buf) {
    int k0 = it << 5;
    #pragma unroll
    for (int q = 0; q < 8; ++q) {
      int idx = (q*128 + tid) * 4;
      int r   = idx >> 5;
      int kc  = idx & 31;
      float4 a4 = *(const float4*)&A[(size_t)(row0 + r) * lda + (k0 + kc)];
      *(uint2*)&As[buf][r*36 + kc] =
          make_uint2(pack_bf16(a4.x, a4.y), pack_bf16(a4.z, a4.w));
    }
  };

  stageB(0, 0);
  stageA(0, 0);
  for (int it = 0; it < ntiles; ++it) {
    const int buf = it & 1;
    const bool has_next = (it + 1) < ntiles;
    if (has_next) { stageB(it + 1, buf ^ 1); stageA(it + 1, buf ^ 1); }
    if (it + 2 < ntiles)
      __builtin_prefetch(&A[(size_t)(row0 + tid) * lda + ((it + 2) << 5)], 0, 0);
    // Wait for THIS tile's async loads only; next tile's 2 stay in flight.
    if (has_next) asm volatile("s_wait_asynccnt 0x2" ::: "memory");
    else          asm volatile("s_wait_asynccnt 0x0" ::: "memory");
    __syncthreads();

    // Fragments per CDNA5 16-bit WMMA layouts (wave32):
    // element pair i packs (k,k+1), k = 2*i + (i>=4 ? 8:0) + (lane>=16 ? 8:0).
    union Frag { unsigned int u32[8]; v16bf v; } af[2], bf[4];
    const int arow = wid*32 + (lane & 15);
    const int ncol = lane & 15;
    const int kx   = (lane >= 16) ? 8 : 0;
    #pragma unroll
    for (int i = 0; i < 8; ++i) {
      int koff = 2*i + ((i >= 4) ? 8 : 0) + kx;
      af[0].u32[i] = *(const unsigned int*)&As[buf][(arow     )*36 + koff];
      af[1].u32[i] = *(const unsigned int*)&As[buf][(arow + 16)*36 + koff];
      #pragma unroll
      for (int j = 0; j < 4; ++j)
        bf[j].u32[i] = *(const unsigned int*)&Bs[buf][(ncol + j*16)*40 + koff];
    }
    #pragma unroll
    for (int j = 0; j < 4; ++j)
      #pragma unroll
      for (int h = 0; h < 2; ++h)
        acc[j*2 + h] = __builtin_amdgcn_wmma_f32_16x16x32_bf16(
            false, af[h].v, false, bf[j].v, (short)0, acc[j*2 + h], false, false);
    __syncthreads();
  }

  // C/D layout: VGPR r -> M = r (lanes 0-15) or r+8 (lanes 16-31); N = lane%16.
  const int cbase = col0 + (lane & 15);
  #pragma unroll
  for (int j = 0; j < 4; ++j) {
    int col = cbase + j*16;
    if (col >= Ncols) continue;
    float bv = (EPI >= 1) ? bias[col] : 0.0f;
    #pragma unroll
    for (int h = 0; h < 2; ++h) {
      int rbase = row0 + wid*32 + h*16 + ((lane >= 16) ? 8 : 0);
      #pragma unroll
      for (int r = 0; r < 8; ++r) {
        float v = acc[j*2 + h][r];
        if (EPI == 1)      v = v + bv;
        else if (EPI == 2) v = geluf_(v + bv);
        else if (EPI == 3) v = softplusf_(v + bv);
        if (mask) v *= mask[rbase + r];
        C[(size_t)(rbase + r) * ldc + col] = v;
      }
    }
  }
}

// ---------------------------------------------------------------------------
// Depthwise causal conv (DCONV=4) over T + bias + SiLU.
// Input xc lives interleaved in xz (row stride 2048, cols 0..1023).
// ---------------------------------------------------------------------------
__global__ void conv_silu_kernel(const float* __restrict__ xz, const float* __restrict__ cw,
                                 const float* __restrict__ cb, float* __restrict__ xc)
{
  int i  = blockIdx.x * blockDim.x + threadIdx.x;   // over B*T*DIN
  int d  = i & (DIN_ - 1);
  int bt = i >> 10;
  int t  = bt & (T_ - 1);
  float acc = cb[d];
  #pragma unroll
  for (int j = 0; j < DCONV_; ++j) {
    int tt = t - (DCONV_ - 1) + j;
    if (tt >= 0) acc += cw[d*DCONV_ + j] * xz[(size_t)(bt + (tt - t)) * (2*DIN_) + d];
  }
  xc[i] = siluf_(acc);
}

// y *= silu(z)  (z = xz cols 1024..2047)
__global__ void ymul_silu_z(const float* __restrict__ xz, float* __restrict__ y)
{
  int i  = blockIdx.x * blockDim.x + threadIdx.x;
  int d  = i & (DIN_ - 1);
  int bt = i >> 10;
  y[i] *= siluf_(xz[(size_t)bt * (2*DIN_) + DIN_ + d]);
}

// ---------------------------------------------------------------------------
// Selective scan: one wave per (b,d). N=64 states -> 2 per lane (n=lane, lane+32).
// Cross-lane y-reduction each step via shfl_xor (wave32 = 5 steps).
// ---------------------------------------------------------------------------
__global__ __launch_bounds__(256)
void scan_kernel(const float* __restrict__ xc, const float* __restrict__ dt,
                 const float* __restrict__ xp, const float* __restrict__ A_log,
                 const float* __restrict__ Dp, float* __restrict__ y)
{
  int gw   = (blockIdx.x * blockDim.x + threadIdx.x) >> 5;
  int lane = threadIdx.x & 31;
  int b = gw >> 10;
  int d = gw & (DIN_ - 1);
  float Aa = -__expf(A_log[d*N_ + lane]);
  float Ab = -__expf(A_log[d*N_ + lane + 32]);
  float Dd = Dp[d];
  float ha = 0.f, hb = 0.f;
  for (int t = 0; t < T_; ++t) {
    size_t bx = (size_t)(b*T_ + t)*DIN_ + d;
    float x   = xc[bx];
    float dtv = dt[bx];
    size_t xb = (size_t)(b*T_ + t)*XPW_;
    float Ba = xp[xb + DTR_ + lane],      Bb = xp[xb + DTR_ + lane + 32];
    float Ca = xp[xb + DTR_ + N_ + lane], Cb = xp[xb + DTR_ + N_ + lane + 32];
    float dx = dtv * x;
    ha = ha * __expf(dtv*Aa) + dx*Ba;
    hb = hb * __expf(dtv*Ab) + dx*Bb;
    float p = ha*Ca + hb*Cb;
    #pragma unroll
    for (int off = 16; off > 0; off >>= 1) p += __shfl_xor(p, off, 32);
    if (lane == 0) y[bx] = p + x*Dd;
  }
}

// other = concat(moutA, moutB) over feature dim (512 + 512 -> 1024)
__global__ void concat2(const float* __restrict__ a, const float* __restrict__ b,
                        float* __restrict__ out)
{
  int i  = blockIdx.x * blockDim.x + threadIdx.x;   // over B*T*1024
  int d  = i & (DIN_ - 1);
  int bt = i >> 10;
  out[i] = (d < H_) ? a[(size_t)bt*H_ + d] : b[(size_t)bt*H_ + d - H_];
}

// ---------------------------------------------------------------------------
// enh = gate*cross + mout + seq; updated = enh*mask; out = LN(seq+updated)*mask
// One block per (b,t); 256 threads, 2 features each (H=512).
// ---------------------------------------------------------------------------
__global__ __launch_bounds__(256)
void enhance_ln(const float* __restrict__ seq, const float* __restrict__ mout,
                const float* __restrict__ cross, const float* __restrict__ gates, int m,
                const float* __restrict__ mask, const float* __restrict__ lng,
                const float* __restrict__ lnb, float* __restrict__ out)
{
  __shared__ float s1s[8], s2s[8], stat[2];
  int bt = blockIdx.x;
  int b  = bt >> 8;
  float g  = gates[b*M_ + m];
  float mk = mask[bt];
  int h0 = threadIdx.x;
  size_t base = (size_t)bt * H_;
  float s0 = seq[base + h0];
  float v0 = s0 + (g*cross[base + h0] + mout[base + h0] + s0) * mk;
  float s1v = seq[base + h0 + 256];
  float v1 = s1v + (g*cross[base + h0 + 256] + mout[base + h0 + 256] + s1v) * mk;

  float s1 = v0 + v1, s2 = v0*v0 + v1*v1;
  #pragma unroll
  for (int off = 16; off > 0; off >>= 1) {
    s1 += __shfl_xor(s1, off, 32);
    s2 += __shfl_xor(s2, off, 32);
  }
  int wid = threadIdx.x >> 5, lane = threadIdx.x & 31;
  if (lane == 0) { s1s[wid] = s1; s2s[wid] = s2; }
  __syncthreads();
  if (threadIdx.x == 0) {
    float t1 = 0.f, t2 = 0.f;
    for (int i = 0; i < 8; ++i) { t1 += s1s[i]; t2 += s2s[i]; }
    float mu  = t1 / (float)H_;
    float var = t2 / (float)H_ - mu*mu;
    stat[0] = mu; stat[1] = rsqrtf(var + EPS_);
  }
  __syncthreads();
  float mu = stat[0], rv = stat[1];
  out[base + h0]       = ((v0 - mu)*rv*lng[h0]       + lnb[h0])       * mk;
  out[base + h0 + 256] = ((v1 - mu)*rv*lng[h0 + 256] + lnb[h0 + 256]) * mk;
}

// ---------------------------------------------------------------------------
// Gating path (tiny): prior normalization, tanh hidden, sigmoid gates.
// ---------------------------------------------------------------------------
__global__ __launch_bounds__(256)
void prior_norm_kernel(const float* __restrict__ prior, float* __restrict__ pn)
{
  __shared__ float ss[8], sc[1];
  int b = blockIdx.x;
  float s = 0.f;
  for (int p = threadIdx.x; p < P_; p += 256) { float v = prior[b*P_ + p]; s += v*v; }
  #pragma unroll
  for (int off = 16; off > 0; off >>= 1) s += __shfl_xor(s, off, 32);
  int wid = threadIdx.x >> 5, lane = threadIdx.x & 31;
  if (lane == 0) ss[wid] = s;
  __syncthreads();
  if (threadIdx.x == 0) {
    float t = 0.f; for (int i = 0; i < 8; ++i) t += ss[i];
    float n = fmaxf(sqrtf(t), 1e-6f);
    sc[0] = 1.0f / n;
  }
  __syncthreads();
  float inv = sc[0];
  for (int p = threadIdx.x; p < P_; p += 256) pn[b*P_ + p] = prior[b*P_ + p] * inv;
}

__global__ void gate_hidden_kernel(const float* __restrict__ pn, const float* __restrict__ Wg1,
                                   const float* __restrict__ bg1, float* __restrict__ hid)
{
  int i = blockIdx.x * blockDim.x + threadIdx.x;
  if (i >= B_*H_) return;
  int b = i / H_, h = i % H_;
  float acc = bg1[h];
  const float* w = Wg1 + (size_t)h * P_;
  const float* p = pn  + (size_t)b * P_;
  for (int k = 0; k < P_; ++k) acc += p[k] * w[k];
  hid[i] = tanhf(acc);
}

__global__ void gate_out_kernel(const float* __restrict__ hid, const float* __restrict__ Wg2,
                                const float* __restrict__ bg2, float* __restrict__ gates)
{
  int i = blockIdx.x * blockDim.x + threadIdx.x;
  if (i >= B_*M_) return;
  int b = i / M_, m = i % M_;
  float acc = bg2[m];
  const float* w  = Wg2 + (size_t)m * H_;
  const float* hh = hid + (size_t)b * H_;
  for (int k = 0; k < H_; ++k) acc += hh[k] * w[k];
  gates[i] = sigmoidf_(acc);
}

// ---------------------------------------------------------------------------
extern "C" void kernel_launch(void* const* d_in, const int* in_sizes, int n_in,
                              void* d_out, int out_size, void* d_ws, size_t ws_size,
                              hipStream_t stream)
{
  (void)in_sizes; (void)n_in; (void)out_size; (void)ws_size;
  const float* text   = (const float*)d_in[0];
  const float* audio  = (const float*)d_in[1];
  const float* vision = (const float*)d_in[2];
  const float* prior  = (const float*)d_in[3];
  const float* mask   = (const float*)d_in[4];
  const float* W_in   = (const float*)d_in[5];
  const float* conv_w = (const float*)d_in[6];
  const float* conv_b = (const float*)d_in[7];
  const float* W_xp   = (const float*)d_in[8];
  const float* W_dt   = (const float*)d_in[9];
  const float* b_dt   = (const float*)d_in[10];
  const float* A_log  = (const float*)d_in[11];
  const float* Dp     = (const float*)d_in[12];
  const float* W_out  = (const float*)d_in[13];
  const float* We1    = (const float*)d_in[14];
  const float* be1    = (const float*)d_in[15];
  const float* We2    = (const float*)d_in[16];
  const float* be2    = (const float*)d_in[17];
  const float* Wg1    = (const float*)d_in[18];
  const float* bg1    = (const float*)d_in[19];
  const float* Wg2    = (const float*)d_in[20];
  const float* bg2    = (const float*)d_in[21];
  const float* ln_g   = (const float*)d_in[22];
  const float* ln_b   = (const float*)d_in[23];
  float* outp_base = (float*)d_out;

  const size_t BTH = (size_t)B_*T_*H_;     // 2,097,152
  const size_t BTD = (size_t)B_*T_*DIN_;   // 4,194,304

  float* ws = (float*)d_ws;
  size_t off = 0;
  float* seqA[M_]; for (int m = 0; m < M_; ++m) { seqA[m] = ws + off; off += BTH; }
  float* seqB[M_]; for (int m = 0; m < M_; ++m) { seqB[m] = ws + off; off += BTH; }
  float* mout[M_]; for (int m = 0; m < M_; ++m) { mout[m] = ws + off; off += BTH; }
  float* xz  = ws + off; off += 2*BTD;
  float* xc  = ws + off; off += BTD;
  float* xp  = ws + off; off += (size_t)B_*T_*XPW_;
  float* dtb = ws + off; off += BTD;
  float* yb  = ws + off; off += BTD;
  float* pn   = ws + off; off += (size_t)B_*P_;
  float* ghid = ws + off; off += (size_t)B_*H_;
  float* gts  = ws + off; off += (size_t)B_*M_;
  unsigned short* wbf = (unsigned short*)(ws + off); off += (size_t)(2*DIN_)*H_/2; // 1M bf16
  // Safe aliases (mamba-stage scratch is dead by the cross-modal stage):
  float* other = xz;
  float* h1    = dtb;
  float* cross = yb;

  hipMemcpyAsync(seqA[0], text,   BTH*sizeof(float), hipMemcpyDeviceToDevice, stream);
  hipMemcpyAsync(seqA[1], audio,  BTH*sizeof(float), hipMemcpyDeviceToDevice, stream);
  hipMemcpyAsync(seqA[2], vision, BTH*sizeof(float), hipMemcpyDeviceToDevice, stream);

  prior_norm_kernel<<<B_, 256, 0, stream>>>(prior, pn);

  // Helper macro: convert a weight slice to bf16 then run the GEMM.
  #define WCVT(SRC, NELEM) \
    wcvt_kernel<<<(((NELEM)/4) + 255)/256, 256, 0, stream>>>((SRC), wbf, (int)((NELEM)/4))

  const int GY = (B_*T_) / 128;   // 32 row-blocks (block tile = 128 rows)

  for (int l = 0; l < L_; ++l) {
    gate_hidden_kernel<<<(B_*H_ + 255)/256, 256, 0, stream>>>(
        pn, Wg1 + (size_t)l*H_*P_, bg1 + (size_t)l*H_, ghid);
    gate_out_kernel<<<1, 64, 0, stream>>>(
        ghid, Wg2 + (size_t)l*M_*H_, bg2 + (size_t)l*M_, gts);

    float* const* src = (l & 1) ? seqB : seqA;
    float* const* dst = (l & 1) ? seqA : seqB;

    // ---- per-modality Mamba blocks ----
    for (int m = 0; m < M_; ++m) {
      size_t lm = (size_t)l*M_ + m;
      // xz = X @ W_in^T                    (4096x512)*(512x2048)
      WCVT(W_in + lm*(size_t)(2*DIN_)*H_, (size_t)(2*DIN_)*H_);
      gemm_wmma_bf16<0><<<dim3(2*DIN_/64, GY), 128, 0, stream>>>(
          src[m], wbf, xz, 2*DIN_, H_, H_, 2*DIN_, nullptr, nullptr);
      // depthwise conv + silu
      conv_silu_kernel<<<(int)(BTD/256), 256, 0, stream>>>(
          xz, conv_w + lm*(size_t)DIN_*DCONV_, conv_b + lm*(size_t)DIN_, xc);
      // xp = xc @ W_xp^T                   (4096x1024)*(1024x160)
      WCVT(W_xp + lm*(size_t)XPW_*DIN_, (size_t)XPW_*DIN_);
      gemm_wmma_bf16<0><<<dim3((XPW_ + 63)/64, GY), 128, 0, stream>>>(
          xc, wbf, xp, XPW_, DIN_, DIN_, XPW_, nullptr, nullptr);
      // dt = softplus(dt_r @ W_dt^T + b_dt)  (4096x32)*(32x1024)
      WCVT(W_dt + lm*(size_t)DIN_*DTR_, (size_t)DIN_*DTR_);
      gemm_wmma_bf16<3><<<dim3(DIN_/64, GY), 128, 0, stream>>>(
          xp, wbf, dtb, DIN_, DTR_, XPW_, DIN_, b_dt + lm*(size_t)DIN_, nullptr);
      // selective scan (wave per (b,d))
      scan_kernel<<<(B_*DIN_*32)/256, 256, 0, stream>>>(
          xc, dtb, xp, A_log + lm*(size_t)DIN_*N_, Dp + lm*(size_t)DIN_, yb);
      // y *= silu(z)
      ymul_silu_z<<<(int)(BTD/256), 256, 0, stream>>>(xz, yb);
      // mout = (y @ W_out^T) * mask        (4096x1024)*(1024x512)
      WCVT(W_out + lm*(size_t)H_*DIN_, (size_t)H_*DIN_);
      gemm_wmma_bf16<0><<<dim3(H_/64, GY), 128, 0, stream>>>(
          yb, wbf, mout[m], H_, DIN_, DIN_, H_, nullptr, mask);
    }

    // ---- cross-modal enhancement + residual layernorm ----
    for (int m = 0; m < M_; ++m) {
      size_t lm = (size_t)l*M_ + m;
      int j1 = (m == 0) ? 1 : 0;
      int j2 = (m == 2) ? 1 : 2;
      concat2<<<(int)(BTD/256), 256, 0, stream>>>(mout[j1], mout[j2], other);
      // h1 = gelu(other @ We1^T + be1)     (4096x1024)*(1024x512)
      WCVT(We1 + lm*(size_t)H_*(2*H_), (size_t)H_*(2*H_));
      gemm_wmma_bf16<2><<<dim3(H_/64, GY), 128, 0, stream>>>(
          other, wbf, h1, H_, 2*H_, 2*H_, H_, be1 + lm*(size_t)H_, nullptr);
      // cross = h1 @ We2^T + be2           (4096x512)*(512x512)
      WCVT(We2 + lm*(size_t)H_*H_, (size_t)H_*H_);
      gemm_wmma_bf16<1><<<dim3(H_/64, GY), 128, 0, stream>>>(
          h1, wbf, cross, H_, H_, H_, H_, be2 + lm*(size_t)H_, nullptr);
      float* op = (l == L_ - 1) ? (outp_base + (size_t)m*BTH) : dst[m];
      enhance_ln<<<B_*T_, 256, 0, stream>>>(
          src[m], mout[m], cross, gts, m, mask,
          ln_g + (size_t)m*H_, ln_b + (size_t)m*H_, op);
    }
  }
  #undef WCVT
}
